// SGU_66236985639652
// MI455X (gfx1250) — matrix operand order repaired and checked by reading, not compiled, not run on hardware
//
#include <hip/hip_runtime.h>

// ---------------------------------------------------------------------------
// SGU (spatial gating unit) for MI455X / gfx1250, wave32 + WMMA bf16.
// Pipeline:
//   1) ln_kernel:      LayerNorm(gate half) -> gate_bf16 [n][d]        (ws)
//   2) gate_tr_kernel: gate_bf16 -> gateT [d][n]                       (ws)
//   3) wcvt_kernel:    W fp32 -> tril-masked bf16 Wb [m][k]            (ws)
//   4) wp_tr_kernel:   proj_w fp32 -> WpT bf16 [n][k]                  (ws)
//   5) spatial_gemm:   G = Wb @ gate ; P = bf16(xh * (G + bias)) -> ws
//   6) proj_gemm:      out = P @ WpT^T + proj_b (f32 out)
// GEMMs: 512 threads (16 waves), 128x128x64 tiles, 32x32 per wave,
// double-buffered LDS fed exclusively by global_load_async_to_lds_b128
// (ASYNCcnt pipeline, s_wait_asynccnt 0x4, last iteration peeled),
// 8x v_wmma_f32_16x16x32_bf16 per barrier pair.
// ---------------------------------------------------------------------------

typedef __attribute__((ext_vector_type(16))) __bf16 v16bf;
typedef __attribute__((ext_vector_type(8)))  float  v8f;

#define SEQ 4096
#define DH  2048
#define DO  1024

#define BM 128
#define BN 128
#define BK 64
#define SA_STRIDE 72   // ushort elems; 144 B rows -> 16B-aligned async chunks
#define SB_STRIDE 72

union BFrag { v16bf v; unsigned int u[8]; };
union AccU  { v8f v; float f[8]; };

__device__ __forceinline__ unsigned short f32_to_bf16(float f) {
  unsigned int u = __float_as_uint(f);
  u += 0x7FFFu + ((u >> 16) & 1u);   // round-to-nearest-even
  return (unsigned short)(u >> 16);
}

__device__ __forceinline__ void async_b128(unsigned int lds_off, const void* g) {
  unsigned long long ga = (unsigned long long)(size_t)g;
  asm volatile("global_load_async_to_lds_b128 %0, %1, off"
               :: "v"(lds_off), "v"(ga) : "memory");
}

// ---------------------------------------------------------------------------
// Kernel 1: LayerNorm over the gate half (scale only, eps 1e-5), bf16 out.
// ---------------------------------------------------------------------------
__global__ void ln_kernel(const float* __restrict__ x,
                          const float* __restrict__ ln_scale,
                          unsigned short* __restrict__ gate_out) {
  const int row = blockIdx.x;
  const float* g = x + (size_t)row * (2 * DH) + DH;   // gate half
  __shared__ float s_sum[256];
  __shared__ float s_sq[256];
  float sum = 0.f, sq = 0.f;
  for (int i = threadIdx.x; i < DH; i += 256) {
    float v = g[i];
    sum += v;
    sq  += v * v;
  }
  s_sum[threadIdx.x] = sum;
  s_sq[threadIdx.x]  = sq;
  __syncthreads();
  for (int off = 128; off > 0; off >>= 1) {
    if (threadIdx.x < off) {
      s_sum[threadIdx.x] += s_sum[threadIdx.x + off];
      s_sq[threadIdx.x]  += s_sq[threadIdx.x + off];
    }
    __syncthreads();
  }
  const float mean = s_sum[0] * (1.0f / DH);
  const float var  = s_sq[0] * (1.0f / DH) - mean * mean;
  const float inv  = rsqrtf(var + 1e-5f);
  for (int i = threadIdx.x; i < DH; i += 256) {
    float v = (g[i] - mean) * inv * ln_scale[i];
    gate_out[(size_t)row * DH + i] = f32_to_bf16(v);
  }
}

// ---------------------------------------------------------------------------
// Kernel 2: W fp32 -> bf16 with branchless tril mask. 8 elems/thread.
// ---------------------------------------------------------------------------
__global__ void wcvt_kernel(const float* __restrict__ W,
                            unsigned short* __restrict__ Wb) {
  const size_t base = ((size_t)blockIdx.x * 256 + threadIdx.x) * 8;
  const int row = (int)(base >> 12);
  const int col = (int)(base & 4095);
  float4 w0 = *(const float4*)(W + base);
  float4 w1 = *(const float4*)(W + base + 4);
  const int rel = row - col;  // element e kept iff e <= rel
  float v[8] = {w0.x, w0.y, w0.z, w0.w, w1.x, w1.y, w1.z, w1.w};
  unsigned int h[8];
#pragma unroll
  for (int e = 0; e < 8; ++e) {
    unsigned int mask = ~(unsigned int)((rel - e) >> 31);  // all-ones iff e<=rel
    h[e] = (unsigned int)f32_to_bf16(v[e]) & mask & 0xFFFFu;
  }
  uint4 p;
  p.x = h[0] | (h[1] << 16);
  p.y = h[2] | (h[3] << 16);
  p.z = h[4] | (h[5] << 16);
  p.w = h[6] | (h[7] << 16);
  *(uint4*)(Wb + base) = p;
}

// ---------------------------------------------------------------------------
// Kernel 3: transpose gate_bf16 [n][d] -> gateT [d][n] (64x64 LDS tiles).
// ---------------------------------------------------------------------------
__global__ void gate_tr_kernel(const unsigned short* __restrict__ g,
                               unsigned short* __restrict__ gT) {
  __shared__ unsigned short t[64][68];
  const int n0 = blockIdx.x * 64;  // 64 blocks
  const int d0 = blockIdx.y * 64;  // 32 blocks
  const int tid = threadIdx.x;     // 256
#pragma unroll
  for (int i = 0; i < 4; ++i) {
    int c = tid + i * 256;
    int rr = c >> 4;          // n
    int cc = (c & 15) * 4;    // d
    uint2 v = *(const uint2*)(g + (size_t)(n0 + rr) * DH + d0 + cc);
    t[rr][cc + 0] = (unsigned short)(v.x & 0xFFFFu);
    t[rr][cc + 1] = (unsigned short)(v.x >> 16);
    t[rr][cc + 2] = (unsigned short)(v.y & 0xFFFFu);
    t[rr][cc + 3] = (unsigned short)(v.y >> 16);
  }
  __syncthreads();
#pragma unroll
  for (int i = 0; i < 4; ++i) {
    int c = tid + i * 256;
    int rr = c >> 4;          // d
    int cc = (c & 15) * 4;    // n
    uint2 v;
    v.x = (unsigned int)t[cc + 0][rr] | ((unsigned int)t[cc + 1][rr] << 16);
    v.y = (unsigned int)t[cc + 2][rr] | ((unsigned int)t[cc + 3][rr] << 16);
    *(uint2*)(gT + (size_t)(d0 + rr) * SEQ + n0 + cc) = v;
  }
}

// ---------------------------------------------------------------------------
// Kernel 4: proj_w [k][n] fp32 -> WpT [n][k] bf16 (64x64 LDS tiles).
// ---------------------------------------------------------------------------
__global__ void wp_tr_kernel(const float* __restrict__ Wp,
                             unsigned short* __restrict__ WpT) {
  __shared__ unsigned short t[64][68];
  const int k0 = blockIdx.x * 64;  // 32 blocks
  const int n0 = blockIdx.y * 64;  // 16 blocks
  const int tid = threadIdx.x;
#pragma unroll
  for (int i = 0; i < 4; ++i) {
    int c = tid + i * 256;
    int rr = c >> 4;          // k
    int cc = (c & 15) * 4;    // n
    float4 v = *(const float4*)(Wp + (size_t)(k0 + rr) * DO + n0 + cc);
    t[rr][cc + 0] = f32_to_bf16(v.x);
    t[rr][cc + 1] = f32_to_bf16(v.y);
    t[rr][cc + 2] = f32_to_bf16(v.z);
    t[rr][cc + 3] = f32_to_bf16(v.w);
  }
  __syncthreads();
#pragma unroll
  for (int i = 0; i < 4; ++i) {
    int c = tid + i * 256;
    int rr = c >> 4;          // n
    int cc = (c & 15) * 4;    // k
    uint2 v;
    v.x = (unsigned int)t[cc + 0][rr] | ((unsigned int)t[cc + 1][rr] << 16);
    v.y = (unsigned int)t[cc + 2][rr] | ((unsigned int)t[cc + 3][rr] << 16);
    *(uint2*)(WpT + (size_t)(n0 + rr) * DH + k0 + cc) = v;
  }
}

// ---------------------------------------------------------------------------
// Fragment loaders per CDNA5 wave32 WMMA layouts (CDNA5 ISA §7.12.2).
// sA is [m][k] row-major, sB is [n][k]: packed pairs -> b128 LDS loads.
// koff selects the K=32 sub-slice within the BK=64 tile.
// ---------------------------------------------------------------------------
__device__ __forceinline__ v16bf load_a_frag(const unsigned short* sA,
                                             int row_base, int koff, int lane) {
  const int half = lane >> 4;
  const int mr   = lane & 15;
  const unsigned int* row =
      (const unsigned int*)(sA + (size_t)(row_base + mr) * SA_STRIDE + koff);
  BFrag f;
  const int kh = half * 4;  // uint units (8 bf16 = 4 uints)
#pragma unroll
  for (int j = 0; j < 4; ++j) {
    f.u[j]     = row[kh + j];      // K = half*8 + 2j, 2j+1
    f.u[4 + j] = row[8 + kh + j];  // K = 16 + half*8 + 2j, 2j+1
  }
  return f.v;
}

__device__ __forceinline__ v16bf load_b_frag(const unsigned short* sB,
                                             int col_base, int koff, int lane) {
  const int half = lane >> 4;
  const int nc   = lane & 15;
  const unsigned int* col =
      (const unsigned int*)(sB + (size_t)(col_base + nc) * SB_STRIDE + koff);
  BFrag f;
#pragma unroll
  for (int j = 0; j < 8; ++j) f.u[j] = col[(j >> 2) * 8 + half * 4 + (j & 3)];  // K = (j/4)*16 + half*8 + 2(j%4)
  return f.v;
}

// ---------------------------------------------------------------------------
// Shared GEMM plumbing: async-copy geometry (4 b128 chunks / thread / tile:
// c = tid + i*512, row = c/8, kchunk = (c%8)*8 over a 128x64 bf16 tile).
// ---------------------------------------------------------------------------
#define GEMM_PROLOGUE(Abase, Astride, Bbase, Bstride, rowA0, rowB0)           \
  const int crow0 = tid >> 3;            /* 0..63  (i=0) */                   \
  const int crow1 = crow0 + 64;          /* 64..127 (i=1) */                  \
  const int ckc   = (tid & 7) * 8;       /* 0,8,...,56 */                     \
  const unsigned short* aSrc0 = (Abase) + (size_t)((rowA0) + crow0) * (Astride) + ckc; \
  const unsigned short* aSrc1 = (Abase) + (size_t)((rowA0) + crow1) * (Astride) + ckc; \
  const unsigned short* bSrc0 = (Bbase) + (size_t)((rowB0) + crow0) * (Bstride) + ckc; \
  const unsigned short* bSrc1 = (Bbase) + (size_t)((rowB0) + crow1) * (Bstride) + ckc; \
  unsigned int aL0[2], aL1[2], bL0[2], bL1[2];                                \
  _Pragma("unroll")                                                           \
  for (int b = 0; b < 2; ++b) {                                               \
    aL0[b] = (unsigned int)(size_t)(&sA[b][crow0 * SA_STRIDE + ckc]);         \
    aL1[b] = (unsigned int)(size_t)(&sA[b][crow1 * SA_STRIDE + ckc]);         \
    bL0[b] = (unsigned int)(size_t)(&sB[b][crow0 * SB_STRIDE + ckc]);         \
    bL1[b] = (unsigned int)(size_t)(&sB[b][crow1 * SB_STRIDE + ckc]);         \
  }

#define GEMM_ISSUE(buf, kt) do {                                              \
    size_t off_ = (size_t)(kt) * BK;                                          \
    async_b128(aL0[buf], aSrc0 + off_);                                       \
    async_b128(aL1[buf], aSrc1 + off_);                                       \
    async_b128(bL0[buf], bSrc0 + off_);                                       \
    async_b128(bL1[buf], bSrc1 + off_);                                       \
  } while (0)

#define GEMM_COMPUTE(buf)                                                     \
  do {                                                                        \
    const unsigned short* cA = sA[buf];                                       \
    const unsigned short* cB = sB[buf];                                       \
    _Pragma("unroll")                                                         \
    for (int ks = 0; ks < BK; ks += 32) {                                     \
      v16bf a[2], b[2];                                                       \
      _Pragma("unroll")                                                       \
      for (int ti = 0; ti < 2; ++ti)                                          \
        a[ti] = load_a_frag(cA, wr * 32 + ti * 16, ks, lane);                 \
      _Pragma("unroll")                                                       \
      for (int tj = 0; tj < 2; ++tj)                                          \
        b[tj] = load_b_frag(cB, wc * 32 + tj * 16, ks, lane);                 \
      _Pragma("unroll")                                                       \
      for (int ti = 0; ti < 2; ++ti)                                          \
        _Pragma("unroll")                                                     \
        for (int tj = 0; tj < 2; ++tj)                                        \
          acc[ti][tj].v = __builtin_amdgcn_wmma_f32_16x16x32_bf16(            \
              false, a[ti], false, b[tj], (short)0, acc[ti][tj].v,            \
              false, false);                                                  \
    }                                                                         \
  } while (0)

// ---------------------------------------------------------------------------
// Kernel 5: G = Wb @ gate (causal pre-masked), fused epilogue
// P = bf16(xh * (G + bias[m])). Double-buffered async-LDS pipeline.
// ---------------------------------------------------------------------------
__global__ void spatial_gemm_kernel(const unsigned short* __restrict__ Wb,   // (SEQ,SEQ) bf16 tril
                                    const unsigned short* __restrict__ gT,   // (DH,SEQ) bf16
                                    const float* __restrict__ x,             // (SEQ,2*DH)
                                    const float* __restrict__ bias,          // (SEQ,1)
                                    unsigned short* __restrict__ P) {        // (SEQ,DH) bf16
  __shared__ unsigned short sA[2][BM * SA_STRIDE];
  __shared__ unsigned short sB[2][BN * SB_STRIDE];

  const int tid  = threadIdx.x;
  const int m0   = blockIdx.y * BM;
  const int d0   = blockIdx.x * BN;
  const int wave = tid >> 5;
  const int lane = tid & 31;
  const int wr   = wave >> 2;
  const int wc   = wave & 3;
  const int half = lane >> 4;
  const int l16  = lane & 15;

  GEMM_PROLOGUE(Wb, SEQ, gT, SEQ, m0, d0)

  AccU acc[2][2];
#pragma unroll
  for (int i = 0; i < 2; ++i)
#pragma unroll
    for (int j = 0; j < 2; ++j)
#pragma unroll
      for (int r = 0; r < 8; ++r) acc[i][j].f[r] = 0.f;

  const int ktiles = (m0 + BM) / BK;  // causal bound: only k <= m contributes
  GEMM_ISSUE(0, 0);
  for (int kt = 0; kt < ktiles - 1; ++kt) {
    GEMM_ISSUE((kt + 1) & 1, kt + 1);
    asm volatile("s_wait_asynccnt 0x4" ::: "memory");  // tile kt landed
    __syncthreads();
    GEMM_COMPUTE(kt & 1);
    __syncthreads();
  }
  asm volatile("s_wait_asynccnt 0x0" ::: "memory");
  __syncthreads();
  GEMM_COMPUTE((ktiles - 1) & 1);

  // Epilogue: P[m,d] = bf16(xh[m,d] * (G + bias[m])).
#pragma unroll
  for (int ti = 0; ti < 2; ++ti) {
#pragma unroll
    for (int tj = 0; tj < 2; ++tj) {
      const int dcol = d0 + wc * 32 + tj * 16 + l16;
#pragma unroll
      for (int r = 0; r < 8; ++r) {
        const int m = m0 + wr * 32 + ti * 16 + half * 8 + r;
        float val = acc[ti][tj].f[r] + bias[m];
        float xh  = x[(size_t)m * (2 * DH) + dcol];  // value half
        P[(size_t)m * DH + dcol] = f32_to_bf16(xh * val);
      }
    }
  }
}

// ---------------------------------------------------------------------------
// Kernel 6: out = P @ WpT^T + proj_b, fp32 out. Same async pipeline.
// ---------------------------------------------------------------------------
__global__ void proj_gemm_kernel(const unsigned short* __restrict__ P,    // (SEQ,DH) bf16
                                 const unsigned short* __restrict__ WpT,  // (DO,DH) bf16
                                 const float* __restrict__ pb,            // (DO,)
                                 float* __restrict__ out) {               // (SEQ,DO)
  __shared__ unsigned short sA[2][BM * SA_STRIDE];
  __shared__ unsigned short sB[2][BN * SB_STRIDE];

  const int tid  = threadIdx.x;
  const int m0   = blockIdx.y * BM;
  const int n0   = blockIdx.x * BN;
  const int wave = tid >> 5;
  const int lane = tid & 31;
  const int wr   = wave >> 2;
  const int wc   = wave & 3;
  const int half = lane >> 4;
  const int l16  = lane & 15;

  GEMM_PROLOGUE(P, DH, WpT, DH, m0, n0)

  AccU acc[2][2];
#pragma unroll
  for (int i = 0; i < 2; ++i)
#pragma unroll
    for (int j = 0; j < 2; ++j)
#pragma unroll
      for (int r = 0; r < 8; ++r) acc[i][j].f[r] = 0.f;

  const int ktiles = DH / BK;  // 32
  GEMM_ISSUE(0, 0);
  for (int kt = 0; kt < ktiles - 1; ++kt) {
    GEMM_ISSUE((kt + 1) & 1, kt + 1);
    asm volatile("s_wait_asynccnt 0x4" ::: "memory");
    __syncthreads();
    GEMM_COMPUTE(kt & 1);
    __syncthreads();
  }
  asm volatile("s_wait_asynccnt 0x0" ::: "memory");
  __syncthreads();
  GEMM_COMPUTE((ktiles - 1) & 1);

#pragma unroll
  for (int ti = 0; ti < 2; ++ti) {
#pragma unroll
    for (int tj = 0; tj < 2; ++tj) {
      const int ncol = n0 + wc * 32 + tj * 16 + l16;
      const float b0 = pb[ncol];
#pragma unroll
      for (int r = 0; r < 8; ++r) {
        const int m = m0 + wr * 32 + ti * 16 + half * 8 + r;
        out[(size_t)m * DO + ncol] = acc[ti][tj].f[r] + b0;
      }
    }
  }
}

// ---------------------------------------------------------------------------
extern "C" void kernel_launch(void* const* d_in, const int* in_sizes, int n_in,
                              void* d_out, int out_size, void* d_ws, size_t ws_size,
                              hipStream_t stream) {
  (void)in_sizes; (void)n_in; (void)out_size; (void)ws_size;
  const float* x        = (const float*)d_in[0];  // (4096, 4096)
  const float* ln_scale = (const float*)d_in[1];  // (2048,)
  const float* W        = (const float*)d_in[2];  // (4096, 4096)
  const float* bias     = (const float*)d_in[3];  // (4096, 1)
  const float* Wp       = (const float*)d_in[4];  // (2048, 1024)
  const float* pb       = (const float*)d_in[5];  // (1024,)
  float* out = (float*)d_out;                     // (4096, 1024)

  // Workspace layout (ushort units). gate region is reused as P after the
  // transpose has consumed it (stream-ordered).
  unsigned short* base      = (unsigned short*)d_ws;
  unsigned short* gate_bf16 = base;                              // 16 MB
  unsigned short* Pbuf      = base;                              // reuse
  unsigned short* gateT     = base + (size_t)SEQ * DH;           // 16 MB
  unsigned short* Wb        = gateT + (size_t)SEQ * DH;          // 32 MB
  unsigned short* WpT       = Wb + (size_t)SEQ * SEQ;            //  4 MB

  ln_kernel<<<SEQ, 256, 0, stream>>>(x, ln_scale, gate_bf16);

  dim3 gtr(SEQ / 64, DH / 64);  // (64, 32)
  gate_tr_kernel<<<gtr, 256, 0, stream>>>(gate_bf16, gateT);

  wcvt_kernel<<<(SEQ * (size_t)SEQ) / (256 * 8), 256, 0, stream>>>(W, Wb);

  dim3 wtr(DH / 64, DO / 64);   // (32, 16)
  wp_tr_kernel<<<wtr, 256, 0, stream>>>(Wp, WpT);

  dim3 g2(DH / BN, SEQ / BM);   // (16, 32)
  spatial_gemm_kernel<<<g2, 512, 0, stream>>>(Wb, gateT, x, bias, Pbuf);

  dim3 g3(DO / BN, SEQ / BM);   // (8, 32)
  proj_gemm_kernel<<<g3, 512, 0, stream>>>(Pbuf, WpT, pb, out);
}
